// NoisyCell_37340445671908
// MI455X (gfx1250) — matrix operand order repaired and checked by reading, not compiled, not run on hardware
//
#include <hip/hip_runtime.h>
#include <hip/hip_bf16.h>

// ---------------------------------------------------------------------------
// 2-layer packed-sequence GRU for MI455X (gfx1250, wave32, WMMA).
// bf16 weights/activations, f32 accumulate via v_wmma_f32_16x16x32_bf16.
// Weights pre-packed fragment-major; per-step GEMMs stream weights L2->LDS
// with GLOBAL_LOAD_ASYNC_TO_LDS_B64 (ASYNCcnt), 4-buffer / 3-deep pipeline,
// one workgroup barrier per K-chunk. The two independent GEMMs of each GRU
// phase are fused into a single 192-block launch.
// ---------------------------------------------------------------------------

typedef __bf16 bf16;
typedef __attribute__((ext_vector_type(16))) __bf16 v16bf;
typedef __attribute__((ext_vector_type(8)))  __bf16 v8bf;
typedef __attribute__((ext_vector_type(8)))  float  v8f;

#define T_STEPS 256
#define BATCH   128
#define E_DIM   512
#define H_DIM   1024
#define NH      3072            // 3*H
#define NBLK32  96              // NH / 32  (blocks per single GEMM, 32 N-cols each)
#define NBUF    4               // async LDS pipeline depth

// ---------------------------------------------------------------------------
// fp32 -> bf16 elementwise convert (for x)
// ---------------------------------------------------------------------------
__global__ __launch_bounds__(256)
void gru_cvt_bf16_kernel(const float* __restrict__ src, bf16* __restrict__ dst, int n) {
    int i = blockIdx.x * 256 + threadIdx.x;
    if (i < n) dst[i] = (bf16)src[i];
}

// ---------------------------------------------------------------------------
// Pack W[N=3072][K] (row major fp32) into WMMA-B-fragment-major bf16,
// grouped per 32-column N block so the GEMM streams contiguous 2KB per
// 32-deep K chunk:
//   dst[((nblk*KT + kt)*1024) + j*512 + lane*16 + e]
//     = W[n][k],  n = nblk*32 + j*16 + (lane&15),
//                 k = kt*32 + (lane>>4)*16 + e
// (contiguous k order per lane half; the A fragment uses the same order).
// ---------------------------------------------------------------------------
__global__ __launch_bounds__(256)
void gru_pack_w_kernel(const float* __restrict__ W, bf16* __restrict__ dst,
                       int K, int total) {
    int o = blockIdx.x * 256 + threadIdx.x;
    if (o >= total) return;
    int e    = o & 15;
    int lane = (o >> 4) & 31;
    int j    = (o >> 9) & 1;
    int rem  = o >> 10;
    int KT   = K >> 5;
    int kt   = rem % KT;
    int nblk = rem / KT;
    int n = nblk * 32 + j * 16 + (lane & 15);
    int k = kt * 32 + (lane >> 4) * 16 + e;
    dst[o] = (bf16)W[(size_t)n * K + k];
}

// ---------------------------------------------------------------------------
// Zero-init hidden state (fp32 state + bf16 activation copies)
// ---------------------------------------------------------------------------
__global__ __launch_bounds__(256)
void gru_init_kernel(float* __restrict__ h0st, float* __restrict__ h1st,
                     bf16* __restrict__ h0bf, bf16* __restrict__ h1bf) {
    int i = blockIdx.x * 256 + threadIdx.x;   // B*H threads
    h0st[i] = 0.f; h1st[i] = 0.f;
    h0bf[i] = (bf16)0.f; h1bf[i] = (bf16)0.f;
}

// ---------------------------------------------------------------------------
// Issue one 2KB cooperative async weight-chunk copy (256 lanes x 8B, b64).
// ---------------------------------------------------------------------------
__device__ __forceinline__
void gru_stage_async(const bf16* wsrc, int kt, unsigned ldsbase, int tid) {
    const bf16* g = wsrc + (size_t)kt * 1024 + tid * 4;
    unsigned dst = ldsbase + (unsigned)((kt & (NBUF - 1)) * 2048 + tid * 8);
    asm volatile("global_load_async_to_lds_b64 %0, %1, off"
                 :: "v"(dst), "v"(g) : "memory");
}

// ---------------------------------------------------------------------------
// GEMM body: C[128 x 3072] = A[128 x K] x Wpk for one 32-column N block.
// 8 wave32s; wave w owns M rows [16w,16w+16). Weights pipelined through
// NBUF LDS buffers via async global->LDS; one barrier per K-chunk:
//   wait(asynccnt<=rem) -> barrier(publish) -> WMMA -> issue(kt+3)
// Overwrite of buf[(kt+3)&3] == buf[(kt-1)&3] is safe: all waves signaled
// this iteration's barrier only after finishing iteration kt-1's reads.
// ---------------------------------------------------------------------------
template<int K>
__device__ __forceinline__
void gru_gemm_body(const bf16* __restrict__ A, int lda,
                   const bf16* __restrict__ Wpk, float* __restrict__ C,
                   int nblk, bf16* ldsB /* [NBUF][1024] bf16 */) {
    constexpr int KT = K >> 5;
    const int tid  = threadIdx.x;
    const int lane = tid & 31;
    const int wave = tid >> 5;                 // 0..7 -> M strip
    const int mrow = wave * 16 + (lane & 15);
    const int ahalf = (lane >> 4) ? 16 : 0;    // A-operand lane K-half

    const bf16* wsrc = Wpk + (size_t)nblk * KT * 1024;
    const unsigned ldsbase = (unsigned)(uintptr_t)ldsB;   // LDS byte offset

    v8f acc0 = {}, acc1 = {};

    // prime the pipeline: chunks 0,1,2 in flight (KT >= 16 > 3)
    gru_stage_async(wsrc, 0, ldsbase, tid);
    gru_stage_async(wsrc, 1, ldsbase, tid);
    gru_stage_async(wsrc, 2, ldsbase, tid);

    #pragma unroll
    for (int kt = 0; kt < KT; ++kt) {
        // Asyncs complete in order: allow the newest min(2, KT-1-kt) to
        // remain in flight; chunk kt is then guaranteed done.
        const int rem = KT - 1 - kt;
        if (rem >= 2)      asm volatile("s_wait_asynccnt 0x2" ::: "memory");
        else if (rem == 1) asm volatile("s_wait_asynccnt 0x1" ::: "memory");
        else               asm volatile("s_wait_asynccnt 0x0" ::: "memory");
        __syncthreads();   // publish buf[kt&3] to all waves

        // A fragment: contiguous k order per lane half (same as packed B)
        //   lanes 0-15 : K[0..15]
        //   lanes16-31 : K[16..31]
        const bf16* ap = A + (size_t)mrow * lda + kt * 32 + ahalf;
        v8bf alo = *(const v8bf*)(ap);
        v8bf ahi = *(const v8bf*)(ap + 8);
        v16bf afrag = __builtin_shufflevector(alo, ahi,
            0, 1, 2, 3, 4, 5, 6, 7, 8, 9, 10, 11, 12, 13, 14, 15);

        const bf16* lb = ldsB + (kt & (NBUF - 1)) * 1024;
        v16bf b0 = *(const v16bf*)(lb + 0 * 512 + lane * 16);
        v16bf b1 = *(const v16bf*)(lb + 1 * 512 + lane * 16);

        acc0 = __builtin_amdgcn_wmma_f32_16x16x32_bf16(false, afrag, false, b0,
                                                       (short)0, acc0, false, false);
        acc1 = __builtin_amdgcn_wmma_f32_16x16x32_bf16(false, afrag, false, b1,
                                                       (short)0, acc1, false, false);

        if (kt + 3 < KT)
            gru_stage_async(wsrc, kt + 3, ldsbase, tid);
    }

    // Epilogue: C 16x16 f32 layout — lane<16: M=r, lane>=16: M=8+r;
    // N = (lane&15) within each 16-wide tile.
    const int mbase = wave * 16 + ((lane >> 4) ? 8 : 0);
    const int ncol  = nblk * 32 + (lane & 15);
    #pragma unroll
    for (int r = 0; r < 8; ++r) {
        C[(size_t)(mbase + r) * NH + ncol +  0] = acc0[r];
        C[(size_t)(mbase + r) * NH + ncol + 16] = acc1[r];
    }
}

// ---------------------------------------------------------------------------
// Fused dual GEMM: blocks [0,96) compute C0 = A0 x W0 (depth K0),
// blocks [96,192) compute C1 = A1 x W1 (depth K1). The two GEMMs of each
// GRU phase are independent, so one launch fills the machine.
// ---------------------------------------------------------------------------
template<int K0, int K1>
__global__ __launch_bounds__(256)
void gru_dualgemm_kernel(const bf16* __restrict__ A0, int lda0,
                         const bf16* __restrict__ W0, float* __restrict__ C0,
                         const bf16* __restrict__ A1, int lda1,
                         const bf16* __restrict__ W1, float* __restrict__ C1) {
    __shared__ alignas(32) bf16 ldsB[NBUF][1024];
    const int b = blockIdx.x;
    if (b < NBLK32) gru_gemm_body<K0>(A0, lda0, W0, C0, b,          &ldsB[0][0]);
    else            gru_gemm_body<K1>(A1, lda1, W1, C1, b - NBLK32, &ldsB[0][0]);
}

// ---------------------------------------------------------------------------
// GRU cell epilogue: gates + packed-sequence masking.
// h_new(unmasked) feeds next layer (actbf); masked value is the carried state.
// ---------------------------------------------------------------------------
__global__ __launch_bounds__(256)
void gru_cell_kernel(const float* __restrict__ gi, const float* __restrict__ gh,
                     const float* __restrict__ bih, const float* __restrict__ bhh,
                     float* __restrict__ hst, bf16* __restrict__ hbf,
                     bf16* __restrict__ actbf,
                     const int* __restrict__ bs, int t) {
    int idx = blockIdx.x * 256 + threadIdx.x;  // B*H threads
    int b = idx >> 10;
    int h = idx & 1023;
    size_t base = (size_t)b * NH + h;
    float ir  = gi[base          ] + bih[h];
    float iz  = gi[base + H_DIM  ] + bih[H_DIM + h];
    float inn = gi[base + 2*H_DIM] + bih[2*H_DIM + h];
    float hr  = gh[base          ] + bhh[h];
    float hz  = gh[base + H_DIM  ] + bhh[H_DIM + h];
    float hn  = gh[base + 2*H_DIM] + bhh[2*H_DIM + h];
    float r = 1.f / (1.f + __expf(-(ir + hr)));
    float z = 1.f / (1.f + __expf(-(iz + hz)));
    float n = tanhf(inn + r * hn);
    float hprev = hst[idx];
    float hnew  = (1.f - z) * n + z * hprev;
    bool  active = b < bs[t];
    float keep = active ? hnew : hprev;
    hst[idx] = keep;
    hbf[idx] = (bf16)keep;
    if (actbf) actbf[idx] = (bf16)hnew;   // pre-mask activation -> next layer
}

// ---------------------------------------------------------------------------
// out[l, j, :] = h_l[unsorted_indices[j], :]
// ---------------------------------------------------------------------------
__global__ __launch_bounds__(256)
void gru_gather_kernel(const float* __restrict__ h0, const float* __restrict__ h1,
                       const int* __restrict__ uidx, float* __restrict__ out) {
    int idx = blockIdx.x * 256 + threadIdx.x;   // 2*B*H threads
    int l   = idx >> 17;                        // B*H = 131072
    int rem = idx & (BATCH * H_DIM - 1);
    int j   = rem >> 10;
    int h   = rem & 1023;
    const float* src = l ? h1 : h0;
    out[idx] = src[(size_t)uidx[j] * H_DIM + h];
}

// ---------------------------------------------------------------------------
extern "C" void kernel_launch(void* const* d_in, const int* in_sizes, int n_in,
                              void* d_out, int out_size, void* d_ws, size_t ws_size,
                              hipStream_t stream) {
    const float* x     = (const float*)d_in[0];
    const float* W_ih0 = (const float*)d_in[1];
    const float* W_hh0 = (const float*)d_in[2];
    const float* b_ih0 = (const float*)d_in[3];
    const float* b_hh0 = (const float*)d_in[4];
    const float* W_ih1 = (const float*)d_in[5];
    const float* W_hh1 = (const float*)d_in[6];
    const float* b_ih1 = (const float*)d_in[7];
    const float* b_hh1 = (const float*)d_in[8];
    const int*   bs    = (const int*)d_in[9];
    const int*   uidx  = (const int*)d_in[10];
    float* out = (float*)d_out;

    // ---- workspace carve-up (bf16 region first, then fp32) ----
    char* ws = (char*)d_ws;
    size_t off = 0;
    auto alloc_bf = [&](size_t elems) { bf16* p = (bf16*)(ws + off); off += elems * sizeof(bf16); return p; };
    bf16* x_bf     = alloc_bf((size_t)T_STEPS * BATCH * E_DIM);  // 32 MB
    bf16* Wih0_pk  = alloc_bf((size_t)NH * E_DIM);
    bf16* Whh0_pk  = alloc_bf((size_t)NH * H_DIM);
    bf16* Wih1_pk  = alloc_bf((size_t)NH * H_DIM);
    bf16* Whh1_pk  = alloc_bf((size_t)NH * H_DIM);
    bf16* h0_bf    = alloc_bf((size_t)BATCH * H_DIM);
    bf16* h1_bf    = alloc_bf((size_t)BATCH * H_DIM);
    bf16* hn0_bf   = alloc_bf((size_t)BATCH * H_DIM);
    off = (off + 255) & ~(size_t)255;
    auto alloc_f = [&](size_t elems) { float* p = (float*)(ws + off); off += elems * sizeof(float); return p; };
    float* h0_st = alloc_f((size_t)BATCH * H_DIM);
    float* h1_st = alloc_f((size_t)BATCH * H_DIM);
    float* gi0   = alloc_f((size_t)BATCH * NH);
    float* gh0   = alloc_f((size_t)BATCH * NH);
    float* gi1   = alloc_f((size_t)BATCH * NH);
    float* gh1   = alloc_f((size_t)BATCH * NH);

    // ---- one-time prep ----
    {
        int n = T_STEPS * BATCH * E_DIM;
        gru_cvt_bf16_kernel<<<n / 256, 256, 0, stream>>>(x, x_bf, n);
    }
    {
        int tot = NH * E_DIM;
        gru_pack_w_kernel<<<tot / 256, 256, 0, stream>>>(W_ih0, Wih0_pk, E_DIM, tot);
    }
    {
        int tot = NH * H_DIM;
        gru_pack_w_kernel<<<tot / 256, 256, 0, stream>>>(W_hh0, Whh0_pk, H_DIM, tot);
        gru_pack_w_kernel<<<tot / 256, 256, 0, stream>>>(W_ih1, Wih1_pk, H_DIM, tot);
        gru_pack_w_kernel<<<tot / 256, 256, 0, stream>>>(W_hh1, Whh1_pk, H_DIM, tot);
    }
    gru_init_kernel<<<(BATCH * H_DIM) / 256, 256, 0, stream>>>(h0_st, h1_st, h0_bf, h1_bf);

    const int cellGrid = (BATCH * H_DIM) / 256;

    // ---- sequential scan over timesteps (4 launches per step) ----
    for (int t = 0; t < T_STEPS; ++t) {
        const bf16* xt = x_bf + (size_t)t * BATCH * E_DIM;
        // layer 0: gi0 = xt*Wih0^T  ||  gh0 = h0*Whh0^T   (independent -> fused)
        gru_dualgemm_kernel<E_DIM, H_DIM><<<2 * NBLK32, 256, 0, stream>>>(
            xt, E_DIM, Wih0_pk, gi0,
            h0_bf, H_DIM, Whh0_pk, gh0);
        gru_cell_kernel<<<cellGrid, 256, 0, stream>>>(gi0, gh0, b_ih0, b_hh0,
                                                      h0_st, h0_bf, hn0_bf, bs, t);
        // layer 1: gi1 = hn0*Wih1^T ||  gh1 = h1*Whh1^T   (independent -> fused)
        gru_dualgemm_kernel<H_DIM, H_DIM><<<2 * NBLK32, 256, 0, stream>>>(
            hn0_bf, H_DIM, Wih1_pk, gi1,
            h1_bf, H_DIM, Whh1_pk, gh1);
        gru_cell_kernel<<<cellGrid, 256, 0, stream>>>(gi1, gh1, b_ih1, b_hh1,
                                                      h1_st, h1_bf, nullptr, bs, t);
    }

    // ---- unsorted_indices gather ----
    gru_gather_kernel<<<(2 * BATCH * H_DIM) / 256, 256, 0, stream>>>(h0_st, h1_st, uidx, out);
}
